// TAG2_40132174414159
// MI455X (gfx1250) — matrix operand order repaired and checked by reading, compiled, not run on hardware
//
#include <hip/hip_runtime.h>
#include <hip/hip_bf16.h>

typedef __attribute__((ext_vector_type(16))) _Float16 v16h;
typedef __attribute__((ext_vector_type(8)))  _Float16 v8h;
typedef __attribute__((ext_vector_type(4)))  _Float16 v4h;
typedef __attribute__((ext_vector_type(8)))  float    v8f;

// ---------------- utility kernels ----------------

__global__ void tag_zero_f32(float* __restrict__ p, int n) {
    int t = blockIdx.x * blockDim.x + threadIdx.x;
    if (t < n) p[t] = 0.0f;
}

__global__ void tag_cvt_f16(const float* __restrict__ s, _Float16* __restrict__ d, int n) {
    int t = blockIdx.x * blockDim.x + threadIdx.x;
    if (t < n) d[t] = (_Float16)s[t];
}

// deg_out[src[e]] += w[e];  deg_in[dst[e]] += w[e]
__global__ void tag_deg_kernel(const float* __restrict__ w,
                               const int* __restrict__ src,
                               const int* __restrict__ dst,
                               float* __restrict__ deg_out,
                               float* __restrict__ deg_in, int E) {
    int e = blockIdx.x * blockDim.x + threadIdx.x;
    if (e >= E) return;
    float we = w[e];
    atomicAdd(&deg_out[src[e]], we);
    atomicAdd(&deg_in[dst[e]], we);
}

// nw = w / sqrt(max(deg_out[src]*deg_in[dst], 1e-12))
__global__ void tag_norm_kernel(const float* __restrict__ w,
                                const int* __restrict__ src,
                                const int* __restrict__ dst,
                                const float* __restrict__ deg_out,
                                const float* __restrict__ deg_in,
                                float* __restrict__ nw, int E) {
    int e = blockIdx.x * blockDim.x + threadIdx.x;
    if (e >= E) return;
    float d = deg_out[src[e]] * deg_in[dst[e]];
    d = fmaxf(d, 1e-12f);
    nw[e] = w[e] * rsqrtf(d);
}

// one wave per edge, 4 floats per lane: out[dst] += nw * fin[src]
__global__ void tag_hop_kernel(const float* __restrict__ fin,
                               const float* __restrict__ nw,
                               const int* __restrict__ src,
                               const int* __restrict__ dst,
                               float* __restrict__ out, int E) {
    int t = blockIdx.x * blockDim.x + threadIdx.x;
    int e = t >> 5;
    if (e >= E) return;
    int lane = t & 31;
    int s = src[e], d = dst[e];
    float wgt = nw[e];
    float4 v = *(const float4*)(fin + (size_t)s * 128 + lane * 4);
    float* po = out + (size_t)d * 128 + lane * 4;
    atomicAdd(po + 0, v.x * wgt);
    atomicAdd(po + 1, v.y * wgt);
    atomicAdd(po + 2, v.z * wgt);
    atomicAdd(po + 3, v.w * wgt);
}

// fc16[N,384] = concat(f0, h1, h2) in fp16
__global__ void tag_concat_f16(const float* __restrict__ f0,
                               const float* __restrict__ h1,
                               const float* __restrict__ h2,
                               _Float16* __restrict__ fc, int total) {
    int t = blockIdx.x * blockDim.x + threadIdx.x;
    if (t >= total) return;
    int row = t / 384;
    int c = t - row * 384;
    float v;
    if (c < 128)       v = f0[(size_t)row * 128 + c];
    else if (c < 256)  v = h1[(size_t)row * 128 + (c - 128)];
    else               v = h2[(size_t)row * 128 + (c - 256)];
    fc[t] = (_Float16)v;
}

// Z[idx,:] = h[a]*h[b] in fp16; rows [0,P)=pos, [P,2P)=neg. One wave per row.
__global__ void tag_build_z(const float* __restrict__ h,
                            const int* __restrict__ ps, const int* __restrict__ pd,
                            const int* __restrict__ ns, const int* __restrict__ nd,
                            _Float16* __restrict__ Z, int P) {
    int t = blockIdx.x * blockDim.x + threadIdx.x;
    int idx = t >> 5;
    if (idx >= 2 * P) return;
    int lane = t & 31;
    int a, b;
    if (idx < P) { a = ps[idx]; b = pd[idx]; }
    else         { a = ns[idx - P]; b = nd[idx - P]; }
    float4 va = *(const float4*)(h + (size_t)a * 128 + lane * 4);
    float4 vb = *(const float4*)(h + (size_t)b * 128 + lane * 4);
    v4h z;
    z[0] = (_Float16)(va.x * vb.x);
    z[1] = (_Float16)(va.y * vb.y);
    z[2] = (_Float16)(va.z * vb.z);
    z[3] = (_Float16)(va.w * vb.w);
    *(v4h*)(Z + (size_t)idx * 128 + lane * 4) = z;
}

// ---------------- WMMA GEMM (register-blocked 16x128 strip per wave) ----------------
// C[M,128] = A[M,K] @ W[128,K]^T + bias, optional relu.
// K is a template constant so every load is a compile-time immediate offset off a
// single base address (no per-iteration 64-bit VALU address math -> no WAR v_nops).
// Per 32-wide K block: load A fragment once + ALL 8 B fragments, then 8 WMMAs,
// letting the scheduler clause the loads and stagger the s_wait_loadcnt.
template <int K>
__global__ void tag_wmma_gemm(const _Float16* __restrict__ A,
                              const _Float16* __restrict__ Bw,   // [128, K] row-major (= W)
                              const float* __restrict__ bias,
                              float* __restrict__ Cf,
                              _Float16* __restrict__ Ch,
                              int M, int relu) {
    int wave = threadIdx.x >> 5;
    int lane = threadIdx.x & 31;
    int mt = blockIdx.x * (blockDim.x >> 5) + wave;
    int m0 = mt * 16;
    if (m0 >= M) return;      // whole-wave uniform exit: EXEC all-ones inside
    int mrow = lane & 15;
    int half = lane >> 4;

    v8f acc[8];
#pragma unroll
    for (int nt = 0; nt < 8; ++nt) acc[nt] = (v8f){};

    // A fragment rows: lanes 0-15 hold K {kb+0..7, kb+16..23}; lanes 16-31 {kb+8..15, kb+24..31}
    const _Float16* Arow = A  + (size_t)(m0 + mrow) * K + half * 8;
    // B fragment (col n=mrow of tile nt): lanes 0-15 hold K {kb+0..7, kb+16..23}; lanes 16-31 {kb+8..15, kb+24..31}
    const _Float16* Brow = Bw + (size_t)mrow * K + half * 8;

#pragma unroll
    for (int kb = 0; kb < K; kb += 32) {
        v8h a0 = *(const v8h*)(Arow + kb);
        v8h a1 = *(const v8h*)(Arow + kb + 16);
        v16h a;
#pragma unroll
        for (int i = 0; i < 8; ++i) { a[i] = a0[i]; a[i + 8] = a1[i]; }

        v16h bf[8];
#pragma unroll
        for (int nt = 0; nt < 8; ++nt) {
            v8h b0 = *(const v8h*)(Brow + nt * 16 * K + kb);
            v8h b1 = *(const v8h*)(Brow + nt * 16 * K + kb + 16);
#pragma unroll
            for (int i = 0; i < 8; ++i) { bf[nt][i] = b0[i]; bf[nt][i + 8] = b1[i]; }
        }

#pragma unroll
        for (int nt = 0; nt < 8; ++nt) {
            acc[nt] = __builtin_amdgcn_wmma_f32_16x16x32_f16(
                false, a, false, bf[nt], (short)0, acc[nt], false, false);
        }
    }

#pragma unroll
    for (int nt = 0; nt < 8; ++nt) {
        int n = nt * 16 + mrow;
        float bv = bias[n];
#pragma unroll
        for (int r = 0; r < 8; ++r) {
            int m = m0 + r + 8 * half;  // C/D layout: VGPR r -> M=r (lanes<16), M=r+8 (lanes>=16)
            float v = acc[nt][r] + bv;
            if (relu) v = fmaxf(v, 0.0f);
            if (Ch) Ch[(size_t)m * 128 + n] = (_Float16)v;
            else    Cf[(size_t)m * 128 + n] = v;
        }
    }
}

// out[row] = dot(Z2[row,:], Wp3) + bp3 ; one wave per row
__global__ void tag_final_dot(const _Float16* __restrict__ Z,
                              const float* __restrict__ w3,
                              const float* __restrict__ b3,
                              float* __restrict__ out, int rows) {
    int t = blockIdx.x * blockDim.x + threadIdx.x;
    int row = t >> 5;
    if (row >= rows) return;
    int lane = t & 31;
    v4h z = *(const v4h*)(Z + (size_t)row * 128 + lane * 4);
    float4 wv = *(const float4*)(w3 + lane * 4);
    float s = (float)z[0] * wv.x + (float)z[1] * wv.y +
              (float)z[2] * wv.z + (float)z[3] * wv.w;
#pragma unroll
    for (int o = 16; o > 0; o >>= 1) s += __shfl_down(s, o, 32);
    if (lane == 0) out[row] = s + b3[0];
}

// ---------------- launcher ----------------

extern "C" void kernel_launch(void* const* d_in, const int* in_sizes, int n_in,
                              void* d_out, int out_size, void* d_ws, size_t ws_size,
                              hipStream_t stream) {
    const float* x   = (const float*)d_in[0];
    const float* w   = (const float*)d_in[1];
    const int*   src = (const int*)d_in[2];
    const int*   dst = (const int*)d_in[3];
    const int*   ps  = (const int*)d_in[4];
    const int*   pd  = (const int*)d_in[5];
    const int*   ns  = (const int*)d_in[6];
    const int*   nd  = (const int*)d_in[7];
    const float* W0  = (const float*)d_in[8];  const float* b0  = (const float*)d_in[9];
    const float* W1  = (const float*)d_in[10]; const float* b1  = (const float*)d_in[11];
    const float* W2  = (const float*)d_in[12]; const float* b2  = (const float*)d_in[13];
    const float* Wp1 = (const float*)d_in[14]; const float* bp1 = (const float*)d_in[15];
    const float* Wp2 = (const float*)d_in[16]; const float* bp2 = (const float*)d_in[17];
    const float* Wp3 = (const float*)d_in[18]; const float* bp3 = (const float*)d_in[19];

    const int N = in_sizes[0] / 128;   // 50000
    const int E = in_sizes[1];         // 800000
    const int P = in_sizes[4];         // 100000
    const int R = 2 * P;               // predictor rows

    // workspace carve-out
    char* wsb = (char*)d_ws;
    size_t off = 0;
    auto carve = [&](size_t bytes) -> void* {
        void* p = wsb + off;
        off += (bytes + 255) & ~(size_t)255;
        return p;
    };
    float*     deg    = (float*)    carve((size_t)2 * N * 4);          // deg_out | deg_in
    float*     nw     = (float*)    carve((size_t)E * 4);
    float*     H12    = (float*)    carve((size_t)2 * N * 128 * 4);    // hop1 | hop2
    float*     Hout   = (float*)    carve((size_t)N * 128 * 4);
    _Float16*  Fc16   = (_Float16*) carve((size_t)N * 384 * 2);
    _Float16*  Zbuf   = (_Float16*) carve((size_t)R * 128 * 2);
    _Float16*  Z2buf  = (_Float16*) carve((size_t)R * 128 * 2);
    _Float16*  W0h    = (_Float16*) carve((size_t)128 * 384 * 2);
    _Float16*  W1h    = (_Float16*) carve((size_t)128 * 384 * 2);
    _Float16*  W2h    = (_Float16*) carve((size_t)128 * 384 * 2);
    _Float16*  Wp1h   = (_Float16*) carve((size_t)128 * 128 * 2);
    _Float16*  Wp2h   = (_Float16*) carve((size_t)128 * 128 * 2);

    float* deg_out = deg;
    float* deg_in  = deg + N;
    float* H1 = H12;
    float* H2 = H12 + (size_t)N * 128;
    float* hfinal = (float*)d_out + (size_t)2 * P;   // d_out = [h_pos | h_neg | h]

    const int TB = 256;
    auto g = [](long long n, int tb) { return (unsigned)((n + tb - 1) / tb); };

    // weight conversions (fp32 -> fp16)
    tag_cvt_f16<<<g(128 * 384, TB), TB, 0, stream>>>(W0, W0h, 128 * 384);
    tag_cvt_f16<<<g(128 * 384, TB), TB, 0, stream>>>(W1, W1h, 128 * 384);
    tag_cvt_f16<<<g(128 * 384, TB), TB, 0, stream>>>(W2, W2h, 128 * 384);
    tag_cvt_f16<<<g(128 * 128, TB), TB, 0, stream>>>(Wp1, Wp1h, 128 * 128);
    tag_cvt_f16<<<g(128 * 128, TB), TB, 0, stream>>>(Wp2, Wp2h, 128 * 128);

    // edge-weight symmetric normalization
    tag_zero_f32<<<g(2 * N, TB), TB, 0, stream>>>(deg, 2 * N);
    tag_deg_kernel<<<g(E, TB), TB, 0, stream>>>(w, src, dst, deg_out, deg_in, E);
    tag_norm_kernel<<<g(E, TB), TB, 0, stream>>>(w, src, dst, deg_out, deg_in, nw, E);

    const _Float16* Wl[3] = {W0h, W1h, W2h};
    const float*    bl[3] = {b0, b1, b2};

    const float* fin = x;
    long long hopThreads = (long long)E * 32;
    long long cvtN = (long long)N * 384;
    // one wave per 16-row strip, 4 waves (128 threads) per block
    unsigned gemmBlocksN = (unsigned)((N / 16 + 3) / 4);
    unsigned gemmBlocksR = (unsigned)((R / 16 + 3) / 4);

    for (int l = 0; l < 3; ++l) {
        tag_zero_f32<<<g((long long)2 * N * 128, TB), TB, 0, stream>>>(H12, 2 * N * 128);
        tag_hop_kernel<<<g(hopThreads, TB), TB, 0, stream>>>(fin, nw, src, dst, H1, E);
        tag_hop_kernel<<<g(hopThreads, TB), TB, 0, stream>>>(H1, nw, src, dst, H2, E);
        tag_concat_f16<<<g(cvtN, TB), TB, 0, stream>>>(fin, H1, H2, Fc16, (int)cvtN);
        float* outp = (l < 2) ? Hout : hfinal;
        tag_wmma_gemm<384><<<gemmBlocksN, 128, 0, stream>>>(
            Fc16, Wl[l], bl[l], outp, (_Float16*)0, N, (l < 2) ? 1 : 0);
        fin = Hout;
    }

    // predictor: Z = h[a]*h[b]; two fp16 WMMA layers; final dot -> h_pos/h_neg
    tag_build_z<<<g((long long)R * 32, TB), TB, 0, stream>>>(hfinal, ps, pd, ns, nd, Zbuf, P);
    tag_wmma_gemm<128><<<gemmBlocksR, 128, 0, stream>>>(
        Zbuf, Wp1h, bp1, (float*)0, Z2buf, R, 1);
    tag_wmma_gemm<128><<<gemmBlocksR, 128, 0, stream>>>(
        Z2buf, Wp2h, bp2, (float*)0, Zbuf, R, 1);
    tag_final_dot<<<g((long long)R * 32, TB), TB, 0, stream>>>(
        Zbuf, Wp3, bp3, (float*)d_out, R);
}
